// SHSA_71992241816075
// MI455X (gfx1250) — matrix operand order repaired and checked by reading, not compiled, not run on hardware
//
#include <hip/hip_runtime.h>

// ---------------------------------------------------------------------------
// SHSA block for MI455X (gfx1250, wave32, WMMA f32_16x16x32_f16).
// Flash-attention fusion: the 8x4096x4096 attention matrix (536 MB) is never
// materialized. All GEMMs run on v_wmma_f32_16x16x32_f16. Intermediates are
// f16 position-major so per-lane loads match the ISA A/B VGPR layouts.
// v-chunk staging in the attention kernel uses the Tensor Data Mover
// (tensor_load_to_lds + s_wait_tensorcnt, double-buffered) when the builtin
// is available; falls back to cooperative global->LDS copies otherwise.
// Workspace use: ~26.2 MB.
// ---------------------------------------------------------------------------

typedef __attribute__((ext_vector_type(16))) _Float16 v16h;
typedef __attribute__((ext_vector_type(8)))  _Float16 v8h;
typedef __attribute__((ext_vector_type(8)))  float    v8f;
typedef __attribute__((ext_vector_type(4)))  unsigned int u32x4;
typedef __attribute__((ext_vector_type(8)))  int      i32x8;
typedef __attribute__((ext_vector_type(4)))  int      i32x4;

#define NB     8
#define NPOS   4096
#define QKDv   16
#define PDIMv  64
#define DIMv   256
#define OCH    96
#define EPSV   1e-5f
#define SCALEV 0.25f   // QKD^-0.5

#if defined(__HIP_DEVICE_COMPILE__) &&                         \
    __has_builtin(__builtin_amdgcn_tensor_load_to_lds) &&      \
    __has_builtin(__builtin_amdgcn_s_wait_tensorcnt)
#define USE_TDM 1
#else
#define USE_TDM 0
#endif

__device__ __forceinline__ v8f vzero8() {
  v8f z;
#pragma unroll
  for (int i = 0; i < 8; i++) z[i] = 0.f;
  return z;
}

__device__ __forceinline__ v8f wmma16(v16h a, v16h b, v8f c) {
  // D = A(16x32 f16) * B(32x16 f16) + C(16x16 f32)
  return __builtin_amdgcn_wmma_f32_16x16x32_f16(
      /*neg_a=*/false, a, /*neg_b=*/false, b,
      /*c_mod=*/(short)0, c, /*reuse_a=*/false, /*reuse_b=*/false);
}

#if USE_TDM
// Issue a TDM load of one contiguous 4 KB v-chunk (2048 f16 elements) into
// LDS. D# built per CDNA5 ISA ch.8: 1-D tile (tile_dim1=0), data_size=2B.
__device__ __forceinline__ void tdm_load_chunk(const _Float16* gptr,
                                               void* lptr) {
  unsigned long long ga = (unsigned long long)(uintptr_t)gptr;
  unsigned int lo = (unsigned int)(uintptr_t)lptr;  // LDS byte offset
  u32x4 g0;
  g0[0] = 1u;                          // count=1 (valid user descriptor)
  g0[1] = lo;                          // lds_addr
  g0[2] = (unsigned int)ga;            // global_addr[31:0]
  g0[3] = (unsigned int)((ga >> 32) & 0x1FFFFFFu) | (2u << 30); // addr[56:32] | type=2
  i32x8 g1;
  g1[0] = 0x10000;                     // data_size=1 (2 bytes), no flags
  g1[1] = 0;                           // atomic_barrier=0, tensor_dim0 lo16=0
  g1[2] = 0x20 | (1 << 16);            // tensor_dim0 hi16 (0x200000 total), tensor_dim1 lo16=1
  g1[3] = (int)(2048u << 16);          // tensor_dim1 hi16=0, tile_dim0=2048
  g1[4] = 0;                           // tile_dim1=0 (1-D), tile_dim2=0
  g1[5] = 2048;                        // tensor_dim0_stride lo32
  g1[6] = 0;
  g1[7] = 0;
  i32x4 z4;
#pragma unroll
  for (int i = 0; i < 4; i++) z4[i] = 0;
#if __clang_major__ >= 23
  i32x8 z8;
#pragma unroll
  for (int i = 0; i < 8; i++) z8[i] = 0;
  __builtin_amdgcn_tensor_load_to_lds(g0, g1, z4, z4, z8, 0);
#else
  __builtin_amdgcn_tensor_load_to_lds(g0, g1, z4, z4, 0);
#endif
}
#endif

// ---------------------------------------------------------------------------
// Kernel 0: fold BN/scale into f16 weights + f32 biases.
// ---------------------------------------------------------------------------
__global__ __launch_bounds__(256) void prep_weights(
    const float* __restrict__ qkv_w, const float* __restrict__ bg,
    const float* __restrict__ bb, const float* __restrict__ brm,
    const float* __restrict__ brv,
    const float* __restrict__ proj_w, const float* __restrict__ pg,
    const float* __restrict__ pb, const float* __restrict__ prm,
    const float* __restrict__ prv,
    _Float16* __restrict__ wq, float* __restrict__ bq,
    _Float16* __restrict__ wp, float* __restrict__ pinv,
    float* __restrict__ poff)
{
  int gid  = blockIdx.x * 256 + threadIdx.x;
  int nthr = gridDim.x * 256;
  for (int i = gid; i < OCH * PDIMv; i += nthr) {
    int o = i >> 6;
    float inv = bg[o] * rsqrtf(brv[o] + EPSV);
    float sc  = (o < QKDv) ? SCALEV : 1.f;
    wq[i] = (_Float16)(qkv_w[i] * inv * sc);
  }
  for (int o = gid; o < OCH; o += nthr) {
    float inv = bg[o] * rsqrtf(brv[o] + EPSV);
    float off = bb[o] - brm[o] * inv;
    bq[o] = off * ((o < QKDv) ? SCALEV : 1.f);
  }
  for (int i = gid; i < DIMv * DIMv; i += nthr) wp[i] = (_Float16)proj_w[i];
  for (int o = gid; o < DIMv; o += nthr) {
    float inv = pg[o] * rsqrtf(prv[o] + EPSV);
    pinv[o] = inv;
    poff[o] = pb[o] - prm[o] * inv;
  }
}

// ---------------------------------------------------------------------------
// Kernel 1: GroupNorm statistics over x1 = x[:, :64] per batch.
// ---------------------------------------------------------------------------
__global__ __launch_bounds__(256) void gn_stats(
    const float* __restrict__ x, float* __restrict__ mu,
    float* __restrict__ rstd)
{
  __shared__ float ssum[256], ssq[256];
  int b = blockIdx.x, tid = threadIdx.x;
  const float* xb = x + (size_t)b * DIMv * NPOS;   // channels 0..63 first
  float s = 0.f, s2 = 0.f;
  for (int i = tid; i < PDIMv * NPOS; i += 256) {
    float v = xb[i];
    s += v; s2 += v * v;
  }
  ssum[tid] = s; ssq[tid] = s2;
  __syncthreads();
  for (int k = 128; k > 0; k >>= 1) {
    if (tid < k) { ssum[tid] += ssum[tid + k]; ssq[tid] += ssq[tid + k]; }
    __syncthreads();
  }
  if (tid == 0) {
    const float inv_n = 1.f / (float)(PDIMv * NPOS);
    float m   = ssum[0] * inv_n;
    float var = ssq[0] * inv_n - m * m;
    mu[b] = m;
    rstd[b] = rsqrtf(var + EPSV);
  }
}

// ---------------------------------------------------------------------------
// Kernel 2: normalize + transpose to position-major f16.
//   c <  64: x1n_t[b][n][c] = f16((x[c,n]-mu)*rstd*g[c] + beta[c])
//   c >= 64: xc_t [b][n][c] = f16(relu(x[c,n]))   (x2 part of concat)
// 32 positions per block: each thread streams 128 B (one full cacheline) of
// x; 2-byte writes are lane-contiguous.
// ---------------------------------------------------------------------------
__global__ __launch_bounds__(256) void norm_tr(
    const float* __restrict__ x, const float* __restrict__ mu,
    const float* __restrict__ rstd, const float* __restrict__ gn_g,
    const float* __restrict__ gn_b,
    _Float16* __restrict__ x1n_t, _Float16* __restrict__ xc_t)
{
  int b = blockIdx.y;
  int n0 = blockIdx.x * 32;
  int c = threadIdx.x;
  float m = mu[b], r = rstd[b];
  float g = 0.f, be = 0.f;
  if (c < PDIMv) {
    g  = gn_g[c] * r;
    be = gn_b[c] - m * r * gn_g[c];
  }
  const float* src = x + ((size_t)b * DIMv + c) * NPOS + n0;
#pragma unroll
  for (int j = 0; j < 32; j++) {
    float v = src[j];
    int n = n0 + j;
    if (c < PDIMv)
      x1n_t[((size_t)b * NPOS + n) * PDIMv + c] = (_Float16)(v * g + be);
    else
      xc_t[((size_t)b * NPOS + n) * DIMv + c] = (_Float16)fmaxf(v, 0.f);
  }
}

// ---------------------------------------------------------------------------
// Kernel 3: qkv GEMM (96x64 @ 64xN) with folded BN, split-stored as
//   q_t[b][n][16] (pre-scaled by SCALE), k_t[b][n][16], v_t[b][n][64], f16.
// ---------------------------------------------------------------------------
__global__ __launch_bounds__(192) void qkv_gemm(
    const _Float16* __restrict__ x1n_t, const _Float16* __restrict__ wq,
    const float* __restrict__ bq,
    _Float16* __restrict__ q_t, _Float16* __restrict__ k_t,
    _Float16* __restrict__ v_t)
{
  int b = blockIdx.y;
  int n0 = blockIdx.x * 16;
  int wave = threadIdx.x >> 5;
  int lane = threadIdx.x & 31;
  int l16 = lane & 15, h = lane >> 4;
  int o0 = wave * 16;
  int n = n0 + l16;

  v8f acc = vzero8();
#pragma unroll
  for (int c0 = 0; c0 < PDIMv; c0 += 32) {
    const _Float16* wrow = wq + (size_t)(o0 + l16) * PDIMv + c0;
    v8h a0 = *(const v8h*)(wrow + 8 * h);
    v8h a1 = *(const v8h*)(wrow + 16 + 8 * h);
    v16h a;
#pragma unroll
    for (int e = 0; e < 8; e++) { a[e] = a0[e]; a[8 + e] = a1[e]; }
    const _Float16* xr = x1n_t + ((size_t)b * NPOS + n) * PDIMv + c0 + 8 * h;
    v8h b0 = *(const v8h*)xr;
    v8h b1 = *(const v8h*)(xr + 16);
    v16h bx;
#pragma unroll
    for (int e = 0; e < 8; e++) { bx[e] = b0[e]; bx[8 + e] = b1[e]; }
    acc = wmma16(a, bx, acc);
  }

  v8h pk;
#pragma unroll
  for (int v = 0; v < 8; v++) {
    int o = o0 + v + 8 * h;
    pk[v] = (_Float16)(acc[v] + bq[o]);
  }
  _Float16* dst;
  if (o0 == 0)
    dst = q_t + ((size_t)b * NPOS + n) * QKDv + 8 * h;
  else if (o0 == 16)
    dst = k_t + ((size_t)b * NPOS + n) * QKDv + 8 * h;
  else
    dst = v_t + ((size_t)b * NPOS + n) * PDIMv + (o0 - 32) + 8 * h;
  *(v8h*)dst = pk;
}

// ---------------------------------------------------------------------------
// Kernel 4: fused flash attention. Grid (N/128, B); 8 waves x 16 rows each.
// Double-buffered LDS v-chunks (TDM async when available), 2 barriers/chunk.
// Per 32-key chunk: 2 QK^T WMMAs, online softmax, P relayout C->A through
// per-wave LDS (wave-internal DS ordering, no barrier), 4 PV WMMAs.
// ---------------------------------------------------------------------------
__global__ __launch_bounds__(256) void attn_fused(
    const _Float16* __restrict__ q_t, const _Float16* __restrict__ k_t,
    const _Float16* __restrict__ v_t, _Float16* __restrict__ xc_t)
{
  __shared__ _Float16 lds_v[2][32][PDIMv];   // 2 x 4 KB staged v chunks
  __shared__ _Float16 lds_p[8][16][32];      // 8 KB per-wave P tiles

  int b = blockIdx.y;
  int tid = threadIdx.x;
  int wave = tid >> 5, lane = tid & 31;
  int l16 = lane & 15, h = lane >> 4;
  int n0 = blockIdx.x * 128 + wave * 16;
  const int NCHUNK = NPOS / 32;

  // A operand (q rows, SCALE pre-folded), K padded 16->32 with zeros.
  v16h aq;
  {
    v8h q8 = *(const v8h*)(q_t + ((size_t)b * NPOS + n0 + l16) * QKDv + 8 * h);
#pragma unroll
    for (int e = 0; e < 8; e++) { aq[e] = q8[e]; aq[8 + e] = (_Float16)0.f; }
  }

  float m_run[8], l_run[8];
#pragma unroll
  for (int v = 0; v < 8; v++) { m_run[v] = -1e30f; l_run[v] = 0.f; }
  v8f oacc[4];
#pragma unroll
  for (int cc = 0; cc < 4; cc++) oacc[cc] = vzero8();

  const _Float16* vbase = v_t + (size_t)b * NPOS * PDIMv;

  // Prologue: stage chunks 0 and 1.
#if USE_TDM
  if (wave == 0) {
    tdm_load_chunk(vbase, &lds_v[0][0][0]);
    tdm_load_chunk(vbase + 32 * PDIMv, &lds_v[1][0][0]);
  }
#else
  {
    int pos = tid >> 3, part = tid & 7;
#pragma unroll
    for (int cb = 0; cb < 2; cb++)
      *(v8h*)(&lds_v[cb][pos][part * 8]) =
          *(const v8h*)(vbase + (size_t)(cb * 32 + pos) * PDIMv + part * 8);
  }
#endif

  for (int it = 0; it < NCHUNK; ++it) {
    int m0 = it * 32;
    int cur = it & 1;
#if USE_TDM
    if (wave == 0) {
      if (it + 1 < NCHUNK) __builtin_amdgcn_s_wait_tensorcnt((short)1);
      else                 __builtin_amdgcn_s_wait_tensorcnt((short)0);
    }
#endif
    __syncthreads();  // barrier A: lds_v[cur] ready for all waves

    // S = (SCALE*q)^T k for two 16-key subtiles
    v8f s[2];
#pragma unroll
    for (int st = 0; st < 2; st++) {
      v16h bk;
#pragma unroll
      for (int e = 0; e < 16; e++) bk[e] = (_Float16)0.f;
      {  // elements 0-7 carry K=8h+e (real); elements 8-15 are K pad
        const _Float16* kp =
            k_t + ((size_t)b * NPOS + m0 + st * 16 + l16) * QKDv + 8 * h;
        v8h k0 = *(const v8h*)kp;
#pragma unroll
        for (int e = 0; e < 8; e++) { bk[e] = k0[e]; }
      }
      s[st] = wmma16(aq, bk, vzero8());
    }

    // Online softmax per row (row = v + 8h, stats reduced over 16 lanes).
    float alpha[8];
#pragma unroll
    for (int v = 0; v < 8; v++) {
      float smax = fmaxf(s[0][v], s[1][v]);
      for (int msk = 1; msk < 16; msk <<= 1)
        smax = fmaxf(smax, __shfl_xor(smax, msk, 32));
      float mnew = fmaxf(m_run[v], smax);
      float p0 = __expf(s[0][v] - mnew);
      float p1 = __expf(s[1][v] - mnew);
      alpha[v] = __expf(m_run[v] - mnew);
      float ps = p0 + p1;
      for (int msk = 1; msk < 16; msk <<= 1)
        ps += __shfl_xor(ps, msk, 32);
      l_run[v] = l_run[v] * alpha[v] + ps;
      m_run[v] = mnew;
      int row = v + 8 * h;
      lds_p[wave][row][l16]      = (_Float16)p0;
      lds_p[wave][row][16 + l16] = (_Float16)p1;
    }
    // per-wave region: DS ops are in-order within a wave; only stop the
    // compiler from reordering the reload above the stores.
    asm volatile("" ::: "memory");

    // Reload P in A layout: row=l16, element e -> K = 16h+e
    v16h pa;
    {
      const _Float16* pr = &lds_p[wave][l16][0];
      v8h p0 = *(const v8h*)(pr + 16 * h);
      v8h p1 = *(const v8h*)(pr + 16 * h + 8);
#pragma unroll
      for (int e = 0; e < 8; e++) { pa[e] = p0[e]; pa[8 + e] = p1[e]; }
    }

    // O[cc] = O[cc]*alpha + P @ v_chunk  (4 c-tiles of 16)
#pragma unroll
    for (int cc = 0; cc < 4; cc++) {
      v16h bv;
      int c = cc * 16 + l16;
#pragma unroll
      for (int e = 0; e < 16; e++) bv[e] = lds_v[cur][16 * h + e][c];
      v8f cin;
#pragma unroll
      for (int v = 0; v < 8; v++) cin[v] = oacc[cc][v] * alpha[v];
      oacc[cc] = wmma16(pa, bv, cin);
    }

    __syncthreads();  // barrier B: all waves done reading lds_v[cur]
    if (it + 2 < NCHUNK) {
#if USE_TDM
      if (wave == 0)
        tdm_load_chunk(vbase + (size_t)(m0 + 64) * PDIMv, &lds_v[cur][0][0]);
#else
      int pos = tid >> 3, part = tid & 7;
      *(v8h*)(&lds_v[cur][pos][part * 8]) =
          *(const v8h*)(vbase + (size_t)(m0 + 64 + pos) * PDIMv + part * 8);
      if (it + 3 < NCHUNK)  // -> global_prefetch_b8
        __builtin_prefetch(
            vbase + (size_t)(m0 + 96 + pos) * PDIMv + part * 8, 0, 1);
#endif
    }
  }

  // Epilogue: normalize, ReLU, write f16 into concat buffer channels 0..63.
#pragma unroll
  for (int cc = 0; cc < 4; cc++) {
#pragma unroll
    for (int v = 0; v < 8; v++) {
      int n = n0 + v + 8 * h;
      int c = cc * 16 + l16;
      float val = oacc[cc][v] / l_run[v];
      xc_t[((size_t)b * NPOS + n) * DIMv + c] = (_Float16)fmaxf(val, 0.f);
    }
  }
}

// ---------------------------------------------------------------------------
// Kernel 5: projection GEMM y = Wp(256x256) @ relu(xc)(256xN) + BN affine.
// ---------------------------------------------------------------------------
__global__ __launch_bounds__(256) void proj_gemm(
    const _Float16* __restrict__ xc_t, const _Float16* __restrict__ wp,
    const float* __restrict__ pinv, const float* __restrict__ poff,
    float* __restrict__ out)
{
  __shared__ _Float16 lds_x[64][DIMv];   // 32 KB
  int b = blockIdx.y;
  int n0 = blockIdx.x * 64;
  int tid = threadIdx.x;
  int wave = tid >> 5, lane = tid & 31;
  int l16 = lane & 15, h = lane >> 4;

  for (int u = tid; u < 2048; u += 256) {  // 2048 x 16B = 32 KB
    int pos = u >> 4, part = u & 15;
    *(v8h*)(&lds_x[pos][part * 8]) =
        *(const v8h*)(xc_t + ((size_t)b * NPOS + n0 + pos) * DIMv + part * 8);
  }
  __syncthreads();

  int ow = wave * 32;
  v8f acc[2][4];
#pragma unroll
  for (int ot = 0; ot < 2; ot++)
#pragma unroll
    for (int nt = 0; nt < 4; nt++) acc[ot][nt] = vzero8();

  for (int c0 = 0; c0 < DIMv; c0 += 32) {
    v16h a[2];
#pragma unroll
    for (int ot = 0; ot < 2; ot++) {
      const _Float16* wr = wp + (size_t)(ow + ot * 16 + l16) * DIMv + c0;
      v8h a0 = *(const v8h*)(wr + 8 * h);
      v8h a1 = *(const v8h*)(wr + 16 + 8 * h);
#pragma unroll
      for (int e = 0; e < 8; e++) { a[ot][e] = a0[e]; a[ot][8 + e] = a1[e]; }
    }
#pragma unroll
    for (int nt = 0; nt < 4; nt++) {
      const _Float16* xr = &lds_x[nt * 16 + l16][c0 + 8 * h];
      v8h b0 = *(const v8h*)xr;
      v8h b1 = *(const v8h*)(xr + 16);
      v16h bx;
#pragma unroll
      for (int e = 0; e < 8; e++) { bx[e] = b0[e]; bx[8 + e] = b1[e]; }
      acc[0][nt] = wmma16(a[0], bx, acc[0][nt]);
      acc[1][nt] = wmma16(a[1], bx, acc[1][nt]);
    }
  }

#pragma unroll
  for (int ot = 0; ot < 2; ot++) {
#pragma unroll
    for (int v = 0; v < 8; v++) {
      int o = ow + ot * 16 + v + 8 * h;
      float inv = pinv[o], off = poff[o];
#pragma unroll
      for (int nt = 0; nt < 4; nt++) {
        int n = n0 + nt * 16 + l16;
        out[((size_t)b * DIMv + o) * NPOS + n] = acc[ot][nt][v] * inv + off;
      }
    }
  }
}

// ---------------------------------------------------------------------------
// Host launcher. Workspace layout (256B-aligned), total 27,409,408 bytes.
// ---------------------------------------------------------------------------
extern "C" void kernel_launch(void* const* d_in, const int* in_sizes, int n_in,
                              void* d_out, int out_size, void* d_ws,
                              size_t ws_size, hipStream_t stream)
{
  (void)in_sizes; (void)n_in; (void)out_size; (void)ws_size;
  const float* x      = (const float*)d_in[0];
  const float* gn_g   = (const float*)d_in[1];
  const float* gn_b   = (const float*)d_in[2];
  const float* qkv_w  = (const float*)d_in[3];
  const float* qbn_g  = (const float*)d_in[4];
  const float* qbn_b  = (const float*)d_in[5];
  const float* qrm    = (const float*)d_in[6];
  const float* qrv    = (const float*)d_in[7];
  const float* proj_w = (const float*)d_in[8];
  const float* pbn_g  = (const float*)d_in[9];
  const float* pbn_b  = (const float*)d_in[10];
  const float* prm    = (const float*)d_in[11];
  const float* prv    = (const float*)d_in[12];
  float* out = (float*)d_out;

  char* ws = (char*)d_ws;
  float*    gn_mu   = (float*)(ws + 0);
  float*    gn_rstd = (float*)(ws + 256);
  float*    bq      = (float*)(ws + 512);        // 96 f32
  float*    pinv    = (float*)(ws + 1024);       // 256 f32
  float*    poff    = (float*)(ws + 2048);       // 256 f32
  _Float16* wq      = (_Float16*)(ws + 3072);    // 96x64 f16   (12 KB)
  _Float16* wp      = (_Float16*)(ws + 15360);   // 256x256 f16 (128 KB)
  _Float16* x1n_t   = (_Float16*)(ws + 146432);  // 8x4096x64  f16 (4 MB)
  _Float16* q_t     = (_Float16*)(ws + 4340736); // 8x4096x16  f16 (1 MB)
  _Float16* k_t     = (_Float16*)(ws + 5389312); // 8x4096x16  f16 (1 MB)
  _Float16* v_t     = (_Float16*)(ws + 6437888); // 8x4096x64  f16 (4 MB)
  _Float16* xc_t    = (_Float16*)(ws + 10632192);// 8x4096x256 f16 (16 MB)

  prep_weights<<<64, 256, 0, stream>>>(qkv_w, qbn_g, qbn_b, qrm, qrv,
                                       proj_w, pbn_g, pbn_b, prm, prv,
                                       wq, bq, wp, pinv, poff);
  gn_stats<<<NB, 256, 0, stream>>>(x, gn_mu, gn_rstd);
  norm_tr<<<dim3(NPOS / 32, NB), 256, 0, stream>>>(x, gn_mu, gn_rstd, gn_g,
                                                   gn_b, x1n_t, xc_t);
  qkv_gemm<<<dim3(NPOS / 16, NB), 192, 0, stream>>>(x1n_t, wq, bq,
                                                    q_t, k_t, v_t);
  attn_fused<<<dim3(NPOS / 128, NB), 256, 0, stream>>>(q_t, k_t, v_t, xc_t);
  proj_gemm<<<dim3(NPOS / 64, NB), 256, 0, stream>>>(xc_t, wp, pinv, poff,
                                                     out);
}
